// MixtureOfExperts_83846351553186
// MI455X (gfx1250) — matrix operand order repaired and checked by reading, not compiled, not run on hardware
//
#include <hip/hip_runtime.h>
#include <hip/hip_bf16.h>

#define DIM   1024
#define HID   4096
#define NEXP  8
#define NTOK  2048   // 2 * 1024 tokens

typedef __attribute__((ext_vector_type(16))) __bf16 v16bf;
typedef __attribute__((ext_vector_type(8)))  __bf16 v8bf;
typedef __attribute__((ext_vector_type(4)))  __bf16 v4bf;
typedef __attribute__((ext_vector_type(8)))  float  v8f;

__device__ __forceinline__ v8f wmma_bf16(v16bf a, v16bf b, v8f c) {
  // (neg_a, A, neg_b, B, c_mod, C, reuse_a, reuse_b)
  return __builtin_amdgcn_wmma_f32_16x16x32_bf16(false, a, false, b, (short)0, c,
                                                 false, false);
}

// A-fragment (16x32 bf16, row-major source in LDS, row_stride in elements).
// lane m=lane&15, h2=lane>>4; elems 0..15 -> K=h2*16+i.
__device__ __forceinline__ v16bf load_a_lds(const __bf16* base, int row_stride,
                                            int k0, int lane) {
  const int m = lane & 15, h2 = lane >> 4;
  const __bf16* p = base + m * row_stride + k0 + h2 * 16;
  v8bf lo = *(const v8bf*)p;
  v8bf hi = *(const v8bf*)(p + 8);
  v16bf a;
#pragma unroll
  for (int i = 0; i < 8; ++i) { a[i] = lo[i]; a[8 + i] = hi[i]; }
  return a;
}

// B-fragment (32x16 bf16) from row-major f32 weights: B[k][n] = W[n0+n][k0+k].
__device__ __forceinline__ v16bf load_b_gf32(const float* W, long long row_stride,
                                             int n0, int k0, int lane) {
  const int n = lane & 15, h2 = lane >> 4;
  const float* p = W + (long long)(n0 + n) * row_stride + k0 + h2 * 16;
  v16bf b;
#pragma unroll
  for (int i = 0; i < 16; i += 4) {
    const float4 v = *(const float4*)(p + i);
    b[i]     = (__bf16)v.x;
    b[i + 1] = (__bf16)v.y;
    b[i + 2] = (__bf16)v.z;
    b[i + 3] = (__bf16)v.w;
  }
  return b;
}

// B-fragment from pre-converted bf16 weights: 16 contiguous bf16 per lane
// (32B -> two global_load_b128, no conversion VALU).
__device__ __forceinline__ v16bf load_b_gb16(const __bf16* W, long long row_stride,
                                             int n0, int k0, int lane) {
  const int n = lane & 15, h2 = lane >> 4;
  const __bf16* p = W + (long long)(n0 + n) * row_stride + k0 + h2 * 16;
  return *(const v16bf*)p;
}

// ---------------- init: zero output + accumulator header ----------------
__global__ void moe_init_kernel(float* __restrict__ y, int n,
                                unsigned int* __restrict__ hdr) {
  const int i = blockIdx.x * blockDim.x + threadIdx.x;
  if (i < n) y[i] = 0.0f;
  if (i < 32) hdr[i] = 0u;
}

// ---------------- one-shot weight conversion f32 -> bf16 ----------------
__global__ void moe_w2bf16_kernel(const float* __restrict__ W1,
                                  const float* __restrict__ W2,
                                  const float* __restrict__ W3,
                                  __bf16* __restrict__ W1b,
                                  __bf16* __restrict__ W2b,
                                  __bf16* __restrict__ W3b) {
  const long long n4 = (long long)NEXP * HID * DIM / 4;  // float4's per matrix
  const long long stride = (long long)gridDim.x * blockDim.x;
  for (long long i = (long long)blockIdx.x * blockDim.x + threadIdx.x; i < n4;
       i += stride) {
    float4 a = ((const float4*)W1)[i];
    v4bf o1; o1[0]=(__bf16)a.x; o1[1]=(__bf16)a.y; o1[2]=(__bf16)a.z; o1[3]=(__bf16)a.w;
    ((v4bf*)W1b)[i] = o1;
    a = ((const float4*)W2)[i];
    v4bf o2; o2[0]=(__bf16)a.x; o2[1]=(__bf16)a.y; o2[2]=(__bf16)a.z; o2[3]=(__bf16)a.w;
    ((v4bf*)W2b)[i] = o2;
    a = ((const float4*)W3)[i];
    v4bf o3; o3[0]=(__bf16)a.x; o3[1]=(__bf16)a.y; o3[2]=(__bf16)a.z; o3[3]=(__bf16)a.w;
    ((v4bf*)W3b)[i] = o3;
  }
}

// ---------------- router: logits, top-2 gates, losses, compaction -------
__global__ void moe_router_kernel(const float* __restrict__ x,
                                  const float* __restrict__ Wr,
                                  int* __restrict__ cnt,
                                  float* __restrict__ zacc,
                                  float* __restrict__ pacc,
                                  int* __restrict__ tok_list,
                                  float* __restrict__ gate_list) {
  const int lane = threadIdx.x & 31;
  const int t = blockIdx.x * 8 + (threadIdx.x >> 5);  // one wave per token
  const float* xr = x + (long long)t * DIM;

  float part[NEXP];
#pragma unroll
  for (int e = 0; e < NEXP; ++e) part[e] = 0.0f;
  for (int d = lane; d < DIM; d += 32) {
    const float xv = xr[d];
#pragma unroll
    for (int e = 0; e < NEXP; ++e) part[e] += xv * Wr[e * DIM + d];
  }
  float lg[NEXP];
#pragma unroll
  for (int e = 0; e < NEXP; ++e) {
    float v = part[e];
#pragma unroll
    for (int off = 16; off > 0; off >>= 1) v += __shfl_xor(v, off, 32);
    lg[e] = v;
  }
  if (lane == 0) {
    float z = 0.0f;
#pragma unroll
    for (int e = 0; e < NEXP; ++e) z += lg[e] * lg[e];
    atomicAdd(zacc, z);

    float mx = lg[0];
#pragma unroll
    for (int e = 1; e < NEXP; ++e) mx = fmaxf(mx, lg[e]);
    float p[NEXP], se = 0.0f;
#pragma unroll
    for (int e = 0; e < NEXP; ++e) { p[e] = __expf(lg[e] - mx); se += p[e]; }
    const float inv = 1.0f / se;
#pragma unroll
    for (int e = 0; e < NEXP; ++e) atomicAdd(&pacc[e], p[e] * inv);

    int i1 = 0;
#pragma unroll
    for (int e = 1; e < NEXP; ++e) if (lg[e] > lg[i1]) i1 = e;
    int i2 = (i1 == 0) ? 1 : 0;
#pragma unroll
    for (int e = 0; e < NEXP; ++e) if (e != i1 && lg[e] > lg[i2]) i2 = e;

    const float w2r = __expf(lg[i2] - lg[i1]);   // softmax over top-2
    const float s = 1.0f + w2r;
    const float w1 = 1.0f / s, w2 = w2r / s;

    const int p1 = atomicAdd(&cnt[i1], 1);
    tok_list[i1 * NTOK + p1] = t; gate_list[i1 * NTOK + p1] = w1;
    const int p2 = atomicAdd(&cnt[i2], 1);
    tok_list[i2 * NTOK + p2] = t; gate_list[i2 * NTOK + p2] = w2;
  }
}

// ---------------- finalize: scalar losses -------------------------------
__global__ void moe_finalize_kernel(const float* __restrict__ zacc,
                                    const float* __restrict__ pacc,
                                    float* __restrict__ losses) {
  if (threadIdx.x == 0 && blockIdx.x == 0) {
    const float z = zacc[0] * (1.0f / (float)(NTOK * NEXP)) * 0.001f;
    float b = 0.0f;
    for (int e = 0; e < NEXP; ++e) {
      const float d = pacc[e] * (1.0f / (float)NTOK) - (1.0f / (float)NEXP);
      b += d * d;
    }
    losses[0] = z;
    losses[1] = b * (1.0f / (float)NEXP);
  }
}

// ---------------- fused sparse SwiGLU FFN (WMMA bf16) -------------------
// One workgroup = (expert, tile of 16 routed tokens). 8 waves.
// B16: weights pre-converted to bf16 in workspace; else convert f32 on the fly.
template <bool B16>
__global__ __launch_bounds__(256)
void moe_ffn_kernel(const float* __restrict__ x,
                    const float* __restrict__ W1f,
                    const float* __restrict__ W2f,
                    const float* __restrict__ W3f,
                    const __bf16* __restrict__ W1b,
                    const __bf16* __restrict__ W2b,
                    const __bf16* __restrict__ W3b,
                    const int*   __restrict__ cnt,
                    const int*   __restrict__ tok_list,
                    const float* __restrict__ gate_list,
                    float* __restrict__ y) {
  __shared__ alignas(16) __bf16 xs[16][DIM + 8];   // +16B pad -> conflict-free
  __shared__ alignas(16) __bf16 act[16][128 + 8];  // staged a = h * silu(g)
  __shared__ int   toks[16];
  __shared__ float gates[16];

  const int e     = blockIdx.x >> 7;     // 128 tiles per expert
  const int tile  = blockIdx.x & 127;
  const int count = cnt[e];
  if (tile * 16 >= count) return;

  const int tid  = threadIdx.x;
  const int wave = tid >> 5;
  const int lane = tid & 31;

  if (tid < 16) {
    const int i = tile * 16 + tid;
    const bool ok = i < count;
    toks[tid]  = ok ? tok_list[e * NTOK + i] : 0;
    gates[tid] = ok ? gate_list[e * NTOK + i] : 0.0f;  // padded rows add 0
  }
  __syncthreads();

  // Gather 16 token rows of x into LDS, f32 -> bf16.
  for (int i = tid; i < 16 * (DIM / 4); i += 256) {
    const int r = i >> 8;          // DIM/4 == 256 float4 per row
    const int c4 = i & 255;
    const float4 v = *(const float4*)(x + (long long)toks[r] * DIM + c4 * 4);
    __bf16* dst = &xs[r][c4 * 4];
    dst[0] = (__bf16)v.x; dst[1] = (__bf16)v.y;
    dst[2] = (__bf16)v.z; dst[3] = (__bf16)v.w;
  }
  __syncthreads();

  const long long eoff = (long long)e * HID * DIM;
  const float*  W1ef = W1f + eoff;
  const float*  W3ef = W3f + eoff;
  const float*  W2ef = W2f + eoff;
  const __bf16* W1eb = W1b + eoff;
  const __bf16* W3eb = W3b + eoff;
  const __bf16* W2eb = W2b + eoff;

  const v8f vzero = {0.f, 0.f, 0.f, 0.f, 0.f, 0.f, 0.f, 0.f};
  v8f acc[8];                         // wave owns output cols [wave*128, +128)
#pragma unroll
  for (int t = 0; t < 8; ++t) acc[t] = vzero;

  for (int hb = 0; hb < HID; hb += 128) {
    // Prefetch this chunk's W2 K-slab for phase B while phase A computes.
    {
      const int row0 = wave * 128 + lane * 4;
#pragma unroll
      for (int rr = 0; rr < 4; ++rr) {
        if (B16) __builtin_prefetch(W2eb + (long long)(row0 + rr) * HID + hb, 0, 1);
        else     __builtin_prefetch(W2ef + (long long)(row0 + rr) * HID + hb, 0, 1);
      }
    }

    // Phase A: this wave's 16 hidden columns of h and g (K = DIM).
    const int jcol = hb + wave * 16;
    v8f ch = vzero, cg = vzero;
#pragma unroll 2
    for (int k0 = 0; k0 < DIM; k0 += 32) {
      const v16bf af = load_a_lds(&xs[0][0], DIM + 8, k0, lane);
      v16bf b1, b3;
      if (B16) { b1 = load_b_gb16(W1eb, DIM, jcol, k0, lane);
                 b3 = load_b_gb16(W3eb, DIM, jcol, k0, lane); }
      else     { b1 = load_b_gf32(W1ef, DIM, jcol, k0, lane);
                 b3 = load_b_gf32(W3ef, DIM, jcol, k0, lane); }
      ch = wmma_bf16(af, b1, ch);
      cg = wmma_bf16(af, b3, cg);
    }
    {
      const int mb = (lane >> 4) * 8;   // C/D layout: c[r] = C[mb+r][lane&15]
      const int n  = lane & 15;
#pragma unroll
      for (int r = 0; r < 8; ++r) {
        const float g = cg[r];
        // silu(g) = g * sigmoid(g); v_rcp_f32 (~1 ulp) instead of IEEE divide.
        const float sig = __builtin_amdgcn_rcpf(1.0f + __expf(-g));
        act[mb + r][wave * 16 + n] = (__bf16)(ch[r] * g * sig);
      }
    }
    __syncthreads();

    // Phase B: out[:, wave*128 .. +128) += act(16x128) @ W2^T (K = 128).
    v16bf af[4];
#pragma unroll
    for (int k = 0; k < 4; ++k) af[k] = load_a_lds(&act[0][0], 128 + 8, k * 32, lane);
#pragma unroll
    for (int nt = 0; nt < 8; ++nt) {
      const int d0 = wave * 128 + nt * 16;
      v8f c = acc[nt];
#pragma unroll
      for (int k = 0; k < 4; ++k) {
        v16bf b2;
        if (B16) b2 = load_b_gb16(W2eb, HID, d0, hb + k * 32, lane);
        else     b2 = load_b_gf32(W2ef, HID, d0, hb + k * 32, lane);
        c = wmma_bf16(af[k], b2, c);
      }
      acc[nt] = c;
    }
    __syncthreads();
  }

  // Epilogue: y[tok, d] += gate * out (exactly 2 commutative adds per element).
  const int mb = (lane >> 4) * 8;
  const int n  = lane & 15;
#pragma unroll
  for (int nt = 0; nt < 8; ++nt) {
    const int d = wave * 128 + nt * 16 + n;
#pragma unroll
    for (int r = 0; r < 8; ++r) {
      const int m = mb + r;
      const float val = acc[nt][r] * gates[m];
      atomicAdd(&y[(long long)toks[m] * DIM + d], val);
    }
  }
}

extern "C" void kernel_launch(void* const* d_in, const int* in_sizes, int n_in,
                              void* d_out, int out_size, void* d_ws, size_t ws_size,
                              hipStream_t stream) {
  (void)in_sizes; (void)n_in; (void)out_size;
  const float* x  = (const float*)d_in[0];
  const float* Wr = (const float*)d_in[1];
  const float* W1 = (const float*)d_in[2];
  const float* W2 = (const float*)d_in[3];
  const float* W3 = (const float*)d_in[4];
  float* y = (float*)d_out;

  // Workspace layout:
  //   bytes [0 .. 192K):   header (cnt[8], zacc, pacc[8]) + tok/gate lists
  //   bytes [192K .. ):    optional bf16 weight mirrors W1b|W2b|W3b
  int*   ws_i = (int*)d_ws;
  float* ws_f = (float*)d_ws;
  int*   cnt       = ws_i;
  float* zacc      = ws_f + 8;
  float* pacc      = ws_f + 9;
  int*   tok_list  = ws_i + 32;
  float* gate_list = ws_f + 32 + NEXP * NTOK;

  const long long welems = (long long)NEXP * HID * DIM;   // per weight matrix
  __bf16* W1b = (__bf16*)((char*)d_ws + (192 << 10));
  __bf16* W2b = W1b + welems;
  __bf16* W3b = W2b + welems;
  const size_t need_b16 = (size_t)(192 << 10) + (size_t)(3 * welems) * sizeof(__bf16);
  const bool use_b16 = ws_size >= need_b16;   // constant per process: deterministic

  const int n_out = NTOK * DIM + 2;
  moe_init_kernel<<<(n_out + 255) / 256, 256, 0, stream>>>(y, n_out,
                                                           (unsigned int*)d_ws);
  moe_router_kernel<<<NTOK / 8, 256, 0, stream>>>(x, Wr, cnt, zacc, pacc,
                                                  tok_list, gate_list);
  moe_finalize_kernel<<<1, 32, 0, stream>>>(zacc, pacc, y + (long long)NTOK * DIM);

  if (use_b16) {
    moe_w2bf16_kernel<<<4096, 256, 0, stream>>>(W1, W2, W3, W1b, W2b, W3b);
    moe_ffn_kernel<true><<<NEXP * (NTOK / 16), 256, 0, stream>>>(
        x, W1, W2, W3, W1b, W2b, W3b, cnt, tok_list, gate_list, y);
  } else {
    moe_ffn_kernel<false><<<NEXP * (NTOK / 16), 256, 0, stream>>>(
        x, W1, W2, W3, W1b, W2b, W3b, cnt, tok_list, gate_list, y);
  }
}
